// ChannelAttentionRs_2388001817025
// MI455X (gfx1250) — matrix operand order repaired and checked by reading, not compiled, not run on hardware
//
#include <hip/hip_runtime.h>
#include <hip/hip_bf16.h>

typedef __attribute__((ext_vector_type(16))) _Float16 v16h;
typedef __attribute__((ext_vector_type(8)))  float    v8f;

#define B_   4
#define C_   128
#define H_   192
#define W_   192
#define HW_  (H_*W_)          /* 36864 */
#define NH_  2
#define D_   64
#define EPS_IN  1e-5f
#define EPS_NRM 1e-12f
#define KSPLIT_ 36            /* K chunks of 1024 for the qk GEMM */

__device__ __forceinline__ float lrelu(float v) { return v > 0.f ? v : 0.05f * v; }

// ---------------------------------------------------------------------------
// K1: grouped 3x3 conv (groups=C, out=3C) + LeakyReLU + InstanceNorm(affine),
//     scattered directly into q/k/v staging [B][4][D][HW] (chan = stream*2+head).
//     One workgroup per (stream, b, out-channel) plane.
// ---------------------------------------------------------------------------
__global__ void qkv_conv_in_kernel(const float* __restrict__ x,
                                   const float* __restrict__ ir,
                                   const float* __restrict__ w,
                                   const float* __restrict__ bias,
                                   const float* __restrict__ gamma,
                                   const float* __restrict__ beta,
                                   float* __restrict__ qa,
                                   float* __restrict__ ka,
                                   float* __restrict__ va)
{
    const int s  = blockIdx.y;                 // 0 = rgb, 1 = ir
    const int b  = blockIdx.x / (3 * C_);
    const int oc = blockIdx.x % (3 * C_);
    const int ic = oc / 3;                     // group = input channel
    const int e  = oc % 3;                     // 0=q 1=k 2=v
    const int hh = ic / D_;
    const int dd = ic % D_;

    const float* src = (s ? ir : x) + ((size_t)(b * C_ + ic)) * HW_;
    float wk[9];
#pragma unroll
    for (int i = 0; i < 9; ++i) wk[i] = w[oc * 9 + i];
    const float bi = bias[oc];

    float* dstsel = (e == 0) ? qa : (e == 1) ? ka : va;
    float* plane  = dstsel + ((size_t)((b * 4 + (s * 2 + hh)) * D_ + dd)) * HW_;

    float sum = 0.f, ssq = 0.f;
    for (int p = threadIdx.x; p < HW_; p += 256) {
        const int yy = p / W_, xx = p % W_;
        float acc = bi;
#pragma unroll
        for (int ky = 0; ky < 3; ++ky) {
            const int iy = yy + ky - 1;
            if (iy < 0 || iy >= H_) continue;
#pragma unroll
            for (int kx = 0; kx < 3; ++kx) {
                const int ix = xx + kx - 1;
                if (ix < 0 || ix >= W_) continue;
                acc += wk[ky * 3 + kx] * src[iy * W_ + ix];
            }
        }
        acc = lrelu(acc);
        plane[p] = acc;
        sum += acc;
        ssq += acc * acc;
    }

    __shared__ float s1[256], s2[256];
    s1[threadIdx.x] = sum;
    s2[threadIdx.x] = ssq;
    __syncthreads();
    for (int off = 128; off > 0; off >>= 1) {
        if ((int)threadIdx.x < off) {
            s1[threadIdx.x] += s1[threadIdx.x + off];
            s2[threadIdx.x] += s2[threadIdx.x + off];
        }
        __syncthreads();
    }
    const float mu   = s1[0] * (1.f / (float)HW_);
    const float var  = s2[0] * (1.f / (float)HW_) - mu * mu;
    const float rstd = rsqrtf(var + EPS_IN);
    const float g = gamma[oc], be = beta[oc];
    for (int p = threadIdx.x; p < HW_; p += 256)
        plane[p] = (plane[p] - mu) * rstd * g + be;
}

// ---------------------------------------------------------------------------
// K2: q/k/v fusion conv: 3x3 over spatial (D, HW), 4 -> 2 channels, + LeakyReLU.
//     blockIdx.z selects q/k/v. One thread per output element.
// ---------------------------------------------------------------------------
__global__ void fuse_conv_kernel(const float* __restrict__ qa,
                                 const float* __restrict__ ka,
                                 const float* __restrict__ va,
                                 const float* __restrict__ wq, const float* __restrict__ bq,
                                 const float* __restrict__ wk, const float* __restrict__ bk,
                                 const float* __restrict__ wv, const float* __restrict__ bv,
                                 float* __restrict__ qf,
                                 float* __restrict__ kf,
                                 float* __restrict__ vf)
{
    const int which = blockIdx.z;
    const float* in  = which == 0 ? qa : which == 1 ? ka : va;
    const float* w   = which == 0 ? wq : which == 1 ? wk : wv;
    const float* bb  = which == 0 ? bq : which == 1 ? bk : bv;
    float*       out = which == 0 ? qf : which == 1 ? kf : vf;

    const size_t total = (size_t)B_ * NH_ * D_ * HW_;
    size_t idx = (size_t)blockIdx.x * 256 + threadIdx.x;
    if (idx >= total) return;

    const int p  = (int)(idx % HW_);
    const int d  = (int)((idx / HW_) % D_);
    const int oh = (int)((idx / ((size_t)HW_ * D_)) % NH_);
    const int b  = (int)(idx / ((size_t)HW_ * D_ * NH_));

    float acc = bb[oh];
    for (int icc = 0; icc < 2 * NH_; ++icc) {
        const float* plane = in + ((size_t)(b * 4 + icc) * D_) * HW_;
#pragma unroll
        for (int kd = 0; kd < 3; ++kd) {
            const int ddn = d + kd - 1;
            if (ddn < 0 || ddn >= D_) continue;
#pragma unroll
            for (int kp = 0; kp < 3; ++kp) {
                const int pp = p + kp - 1;
                if (pp < 0 || pp >= HW_) continue;
                acc += w[((oh * 4 + icc) * 3 + kd) * 3 + kp] * plane[(size_t)ddn * HW_ + pp];
            }
        }
    }
    out[idx] = lrelu(acc);
}

// ---------------------------------------------------------------------------
// K3: per-row (b,h,d) reciprocal L2 norms over HW for q and k (F.normalize).
// ---------------------------------------------------------------------------
__global__ void rownorm_kernel(const float* __restrict__ qf,
                               const float* __restrict__ kf,
                               float* __restrict__ rnq,
                               float* __restrict__ rnk)
{
    const float* src = blockIdx.y ? kf : qf;
    float*       dst = blockIdx.y ? rnk : rnq;
    const int row = blockIdx.x;                       // 0 .. B*NH*D-1
    const float* r = src + (size_t)row * HW_;
    float ss = 0.f;
    for (int p = threadIdx.x; p < HW_; p += 256) { const float v = r[p]; ss += v * v; }
    __shared__ float sm[256];
    sm[threadIdx.x] = ss;
    __syncthreads();
    for (int off = 128; off > 0; off >>= 1) {
        if ((int)threadIdx.x < off) sm[threadIdx.x] += sm[threadIdx.x + off];
        __syncthreads();
    }
    if (threadIdx.x == 0) dst[row] = 1.f / fmaxf(sqrtf(sm[0]), EPS_NRM);
}

// ---------------------------------------------------------------------------
// K3b: zero the attn accumulator (re-run every call; replay-safe).
// ---------------------------------------------------------------------------
__global__ void zero_kernel(float* __restrict__ p, int n)
{
    const int i = blockIdx.x * 256 + threadIdx.x;
    if (i < n) p[i] = 0.f;
}

// ---------------------------------------------------------------------------
// K4: attn += qn . kn^T  via v_wmma_f32_16x16x32_f16, K split into KSPLIT_
//     chunks accumulated with global_atomic_add_f32. Each wave owns a 32x32
//     register block (2 A-frags x 2 B-frags x 4 accumulators = 4 wmma/K-step),
//     giving 2x in-register reuse of both operands. Grid (KSPLIT_, B*NH),
//     block = 128 (4 waves = four 32x32 quadrants of the 64x64 result).
//     f16 operand layout per CDNA5 ISA 7.12.2 (16-bit A 16x32 / B 32x16).
// ---------------------------------------------------------------------------
__global__ void qk_wmma_kernel(const float* __restrict__ qf,
                               const float* __restrict__ kf,
                               const float* __restrict__ rnq,
                               const float* __restrict__ rnk,
                               float* __restrict__ attn)
{
    const int bh   = blockIdx.y;                 // b*NH + h
    const int wave = threadIdx.x >> 5;           // 0..3
    const int lane = threadIdx.x & 31;
    const int mi = wave >> 1, ni = wave & 1;     // 32x32 quadrant
    const int lm = lane & 15;
    const int hi = lane >> 4;

    const int m0 = mi * 32 + lm,  m1 = m0 + 16;  // A rows of the two M-tiles
    const int n0 = ni * 32 + lm,  n1 = n0 + 16;  // B cols of the two N-tiles
    const float* qrow0 = qf + ((size_t)bh * D_ + m0) * HW_;
    const float* qrow1 = qf + ((size_t)bh * D_ + m1) * HW_;
    const float* krow0 = kf + ((size_t)bh * D_ + n0) * HW_;
    const float* krow1 = kf + ((size_t)bh * D_ + n1) * HW_;
    const float sq0 = rnq[bh * D_ + m0], sq1 = rnq[bh * D_ + m1];
    const float sk0 = rnk[bh * D_ + n0], sk1 = rnk[bh * D_ + n1];
    const int koffA = hi ? 8 : 0;                // A: lanes 16-31 hold K 8-15 / 24-31
    const int koffB = hi ? 8 : 0;                // B: lanes 16-31 hold K 8-15 / 24-31

    const int kbeg = blockIdx.x * (HW_ / KSPLIT_);
    const int kend = kbeg + (HW_ / KSPLIT_);

    v8f c00 = {}, c01 = {}, c10 = {}, c11 = {};
    for (int k0 = kbeg; k0 < kend; k0 += 32) {
        v16h a0, a1, b0, b1;
        {
            alignas(16) float t0[16], t1[16];
            *(float4*)(t0 + 0)  = *(const float4*)(qrow0 + k0 + koffA);
            *(float4*)(t0 + 4)  = *(const float4*)(qrow0 + k0 + koffA + 4);
            *(float4*)(t0 + 8)  = *(const float4*)(qrow0 + k0 + koffA + 16);
            *(float4*)(t0 + 12) = *(const float4*)(qrow0 + k0 + koffA + 20);
            *(float4*)(t1 + 0)  = *(const float4*)(qrow1 + k0 + koffA);
            *(float4*)(t1 + 4)  = *(const float4*)(qrow1 + k0 + koffA + 4);
            *(float4*)(t1 + 8)  = *(const float4*)(qrow1 + k0 + koffA + 16);
            *(float4*)(t1 + 12) = *(const float4*)(qrow1 + k0 + koffA + 20);
#pragma unroll
            for (int j = 0; j < 16; ++j) {
                a0[j] = (_Float16)(t0[j] * sq0);
                a1[j] = (_Float16)(t1[j] * sq1);
            }
        }
        {
            alignas(16) float t0[16], t1[16];
            *(float4*)(t0 + 0)  = *(const float4*)(krow0 + k0 + koffB);
            *(float4*)(t0 + 4)  = *(const float4*)(krow0 + k0 + koffB + 4);
            *(float4*)(t0 + 8)  = *(const float4*)(krow0 + k0 + koffB + 16);
            *(float4*)(t0 + 12) = *(const float4*)(krow0 + k0 + koffB + 20);
            *(float4*)(t1 + 0)  = *(const float4*)(krow1 + k0 + koffB);
            *(float4*)(t1 + 4)  = *(const float4*)(krow1 + k0 + koffB + 4);
            *(float4*)(t1 + 8)  = *(const float4*)(krow1 + k0 + koffB + 16);
            *(float4*)(t1 + 12) = *(const float4*)(krow1 + k0 + koffB + 20);
#pragma unroll
            for (int j = 0; j < 16; ++j) {
                b0[j] = (_Float16)(t0[j] * sk0);
                b1[j] = (_Float16)(t1[j] * sk1);
            }
        }
        c00 = __builtin_amdgcn_wmma_f32_16x16x32_f16(false, a0, false, b0, (short)0, c00, false, false);
        c01 = __builtin_amdgcn_wmma_f32_16x16x32_f16(false, a0, false, b1, (short)0, c01, false, false);
        c10 = __builtin_amdgcn_wmma_f32_16x16x32_f16(false, a1, false, b0, (short)0, c10, false, false);
        c11 = __builtin_amdgcn_wmma_f32_16x16x32_f16(false, a1, false, b1, (short)0, c11, false, false);
    }

    float* arow = attn + (size_t)bh * D_ * D_;
#pragma unroll
    for (int r = 0; r < 8; ++r) {
        const int r0 = mi * 32 + (hi ? 8 + r : r);      // M-tile 0 rows
        const int r1 = r0 + 16;                         // M-tile 1 rows
        const int cA = ni * 32 + lm;                    // N-tile 0 cols
        const int cB = cA + 16;                         // N-tile 1 cols
        atomicAdd(arow + r0 * D_ + cA, c00[r]);
        atomicAdd(arow + r0 * D_ + cB, c01[r]);
        atomicAdd(arow + r1 * D_ + cA, c10[r]);
        atomicAdd(arow + r1 * D_ + cB, c11[r]);
    }
}

// ---------------------------------------------------------------------------
// K5: scale by t then softmax over last axis of attn [B*NH*D rows of 64].
// ---------------------------------------------------------------------------
__global__ void softmax_kernel(float* __restrict__ attn, const float* __restrict__ t)
{
    const int row = blockIdx.x;                 // bh*64 + d
    const int h = (row / D_) % NH_;
    float* r = attn + (size_t)row * D_;
    const float v = r[threadIdx.x] * t[h];
    __shared__ float sm[64];
    sm[threadIdx.x] = v;
    __syncthreads();
    for (int off = 32; off > 0; off >>= 1) {
        if ((int)threadIdx.x < off)
            sm[threadIdx.x] = fmaxf(sm[threadIdx.x], sm[threadIdx.x + off]);
        __syncthreads();
    }
    const float mx = sm[0];
    __syncthreads();
    const float e = __expf(v - mx);
    sm[threadIdx.x] = e;
    __syncthreads();
    for (int off = 32; off > 0; off >>= 1) {
        if ((int)threadIdx.x < off) sm[threadIdx.x] += sm[threadIdx.x + off];
        __syncthreads();
    }
    r[threadIdx.x] = e / sm[0];
}

// ---------------------------------------------------------------------------
// K6: out = attn @ v via WMMA (M=64, K=64, N=HW). One wave owns one 16-wide
//     N-strip: the strided B-fragments of v are loaded ONCE and reused across
//     all four M-tiles (attn A-frags are 128KB total -> L2 hot). Writes the
//     reshaped image [B][C=128][H][W] directly (channel = d*NH + h).
// ---------------------------------------------------------------------------
__global__ void av_wmma_kernel(const float* __restrict__ attn,
                               const float* __restrict__ vf,
                               float* __restrict__ outimg)
{
    const int bh = blockIdx.y;
    const int b = bh / NH_, h = bh % NH_;
    const int wave = threadIdx.x >> 5;
    const int lane = threadIdx.x & 31;
    const int nt = blockIdx.x * 8 + wave;          // 0..2303
    const int lm = lane & 15;
    const int hi = lane >> 4;

    // B fragments: v columns (K strided by HW); loaded once per wave.
    const int n = nt * 16 + lm;
    const float* vcol = vf + (size_t)bh * D_ * HW_ + n;
    const int koffB = hi ? 8 : 0;
    v16h b0, b1;
#pragma unroll
    for (int j = 0; j < 16; ++j) {
        b0[j] = (_Float16)vcol[(size_t)(koffB + (j & 7) + ((j >> 3) << 4)) * HW_];
        b1[j] = (_Float16)vcol[(size_t)(32 + koffB + (j & 7) + ((j >> 3) << 4)) * HW_];
    }

    const int koffA = hi ? 8 : 0;
    for (int mt = 0; mt < 4; ++mt) {
        // A fragments (attn rows are length-64, contiguous).
        const int m = mt * 16 + lm;
        const float* arow = attn + (size_t)bh * D_ * D_ + (size_t)m * D_;
        v16h a0, a1;
        {
            alignas(16) float v0[16], v1[16];
            *(float4*)(v0 + 0)  = *(const float4*)(arow + koffA);
            *(float4*)(v0 + 4)  = *(const float4*)(arow + koffA + 4);
            *(float4*)(v0 + 8)  = *(const float4*)(arow + koffA + 16);
            *(float4*)(v0 + 12) = *(const float4*)(arow + koffA + 20);
            *(float4*)(v1 + 0)  = *(const float4*)(arow + 32 + koffA);
            *(float4*)(v1 + 4)  = *(const float4*)(arow + 32 + koffA + 4);
            *(float4*)(v1 + 8)  = *(const float4*)(arow + 32 + koffA + 16);
            *(float4*)(v1 + 12) = *(const float4*)(arow + 32 + koffA + 20);
#pragma unroll
            for (int j = 0; j < 16; ++j) { a0[j] = (_Float16)v0[j]; a1[j] = (_Float16)v1[j]; }
        }

        v8f c = {};
        c = __builtin_amdgcn_wmma_f32_16x16x32_f16(false, a0, false, b0, (short)0, c, false, false);
        c = __builtin_amdgcn_wmma_f32_16x16x32_f16(false, a1, false, b1, (short)0, c, false, false);

#pragma unroll
        for (int r = 0; r < 8; ++r) {
            const int d = mt * 16 + (hi ? 8 + r : r);
            outimg[((size_t)b * C_ + (d * NH_ + h)) * HW_ + n] = c[r];
        }
    }
}

// ---------------------------------------------------------------------------
// K7: group_fus depthwise 3x3 conv + bias + LeakyReLU -> d_out.
// ---------------------------------------------------------------------------
__global__ void fus_conv_kernel(const float* __restrict__ in,
                                const float* __restrict__ w,
                                const float* __restrict__ bias,
                                float* __restrict__ out)
{
    const size_t total = (size_t)B_ * C_ * HW_;
    size_t idx = (size_t)blockIdx.x * 256 + threadIdx.x;
    if (idx >= total) return;
    const int p  = (int)(idx % HW_);
    const int cc = (int)((idx / HW_) % C_);
    const int b  = (int)(idx / ((size_t)HW_ * C_));
    const int yy = p / W_, xx = p % W_;
    const float* plane = in + ((size_t)(b * C_ + cc)) * HW_;
    __builtin_prefetch(plane + p + W_, 0, 0);      // global_prefetch_b8
    float acc = bias[cc];
#pragma unroll
    for (int ky = 0; ky < 3; ++ky) {
        const int iy = yy + ky - 1;
        if (iy < 0 || iy >= H_) continue;
#pragma unroll
        for (int kx = 0; kx < 3; ++kx) {
            const int ix = xx + kx - 1;
            if (ix < 0 || ix >= W_) continue;
            acc += w[cc * 9 + ky * 3 + kx] * plane[iy * W_ + ix];
        }
    }
    out[idx] = lrelu(acc);
}

// ---------------------------------------------------------------------------
extern "C" void kernel_launch(void* const* d_in, const int* in_sizes, int n_in,
                              void* d_out, int out_size, void* d_ws, size_t ws_size,
                              hipStream_t stream) {
    const float* x      = (const float*)d_in[0];
    const float* ir     = (const float*)d_in[1];
    const float* w_qkv  = (const float*)d_in[2];
    const float* b_qkv  = (const float*)d_in[3];
    const float* gamma  = (const float*)d_in[4];
    const float* beta   = (const float*)d_in[5];
    const float* wq     = (const float*)d_in[6];
    const float* bq     = (const float*)d_in[7];
    const float* wk     = (const float*)d_in[8];
    const float* bk     = (const float*)d_in[9];
    const float* wv     = (const float*)d_in[10];
    const float* bv     = (const float*)d_in[11];
    const float* t      = (const float*)d_in[12];
    const float* w_fus  = (const float*)d_in[13];
    const float* b_fus  = (const float*)d_in[14];
    float* out = (float*)d_out;

    // Workspace carve-up (floats).
    float* ws = (float*)d_ws;
    const size_t szStage = (size_t)B_ * 4 * D_ * HW_;       // 37,748,736
    const size_t szFused = (size_t)B_ * NH_ * D_ * HW_;     // 18,874,368
    size_t o = 0;
    float* qa   = ws + o; o += szStage;
    float* ka   = ws + o; o += szStage;
    float* va   = ws + o; o += szStage;
    float* qf   = ws + o; o += szFused;
    float* kf   = ws + o; o += szFused;
    float* vf   = ws + o; o += szFused;
    float* attn = ws + o; o += (size_t)B_ * NH_ * D_ * D_;  // 32768
    float* rnq  = ws + o; o += (size_t)B_ * NH_ * D_;       // 512
    float* rnk  = ws + o; o += (size_t)B_ * NH_ * D_;       // 512
    float* aimg = ws + o; o += (size_t)B_ * C_ * HW_;       // 18,874,368

    // K1: shared-weight grouped conv + IN for both streams.
    qkv_conv_in_kernel<<<dim3(B_ * 3 * C_, 2), 256, 0, stream>>>(
        x, ir, w_qkv, b_qkv, gamma, beta, qa, ka, va);

    // K2: q/k/v fusion convs (z = 0/1/2).
    {
        const int blocks = (int)((szFused + 255) / 256);
        fuse_conv_kernel<<<dim3(blocks, 1, 3), 256, 0, stream>>>(
            qa, ka, va, wq, bq, wk, bk, wv, bv, qf, kf, vf);
    }

    // K3: reciprocal L2 row norms for q and k;  K3b: zero attn accumulator.
    rownorm_kernel<<<dim3(B_ * NH_ * D_, 2), 256, 0, stream>>>(qf, kf, rnq, rnk);
    {
        const int nAttn = B_ * NH_ * D_ * D_;               // 32768
        zero_kernel<<<dim3((nAttn + 255) / 256), 256, 0, stream>>>(attn, nAttn);
    }

    // K4: qn . kn^T via WMMA, K-split x36, f32-atomic accumulation.
    qk_wmma_kernel<<<dim3(KSPLIT_, B_ * NH_), 128, 0, stream>>>(qf, kf, rnq, rnk, attn);

    // K5: *t then softmax over last axis.
    softmax_kernel<<<dim3(B_ * NH_ * D_), 64, 0, stream>>>(attn, t);

    // K6: attn @ v via WMMA, writing reshaped image layout.
    av_wmma_kernel<<<dim3(288, B_ * NH_), 256, 0, stream>>>(attn, vf, aimg);

    // K7: depthwise fusion conv -> output.
    {
        const size_t total = (size_t)B_ * C_ * HW_;
        const int blocks = (int)((total + 255) / 256);
        fus_conv_kernel<<<dim3(blocks), 256, 0, stream>>>(aimg, w_fus, b_fus, out);
    }
    (void)in_sizes; (void)n_in; (void)out_size; (void)ws_size;
}
